// SparseAttention_2070174236776
// MI455X (gfx1250) — matrix operand order repaired and checked by reading, not compiled, not run on hardware
//
#include <hip/hip_runtime.h>

// Problem constants (from reference): q,k,v (2, 8192, 256) fp32
#define NB     2
#define TT     8192
#define EE     256
#define SCALE  0.125f      // 1/sqrt(256/4)
#define BK     64          // keys per inner iteration
#define QROWS  64          // query rows per block (16 per wave)
#define SK_STR 264         // padded K-tile row stride (halves)
#define SV_STR 72          // padded Vt / P row stride (halves)
#define NIT    (TT / BK)   // 128 KV iterations

typedef __attribute__((ext_vector_type(16))) _Float16 v16h;
typedef __attribute__((ext_vector_type(8)))  _Float16 v8h;
typedef __attribute__((ext_vector_type(8)))  float    v8f;

__device__ __forceinline__ v16h cat16(v8h lo, v8h hi) {
  return __builtin_shufflevector(lo, hi, 0,1,2,3,4,5,6,7,8,9,10,11,12,13,14,15);
}
__device__ __forceinline__ v8f wmma_f16(v16h a, v16h b, v8f c) {
  // D = A*B + C, fp32 accumulate. args: (neg_a, A, neg_b, B, c_mod, C, reuse_a, reuse_b)
  return __builtin_amdgcn_wmma_f32_16x16x32_f16(false, a, false, b, (short)0, c, false, false);
}

// ---- permutation: xp[n'][t'][e'] = x[nn][b*512+l][ee] -------------------
__device__ __forceinline__ size_t perm_src(int np, int tp, int ep) {
  int nn = (ep >> 4) & 1;
  int b  = ep & 15;
  int l  = np * 256 + (tp >> 5);
  int ee = (tp & 31) * 8 + (ep >> 5);
  return (size_t)(nn * TT + b * 512 + l) * EE + ee;
}

__global__ void pack_qk_kernel(const float* __restrict__ q, const float* __restrict__ k,
                               _Float16* __restrict__ qp, _Float16* __restrict__ kp) {
  int idx = blockIdx.x * 256 + threadIdx.x;         // [n'][t'][e'] flat
  int ep = idx & 255;
  int tp = (idx >> 8) & (TT - 1);
  int np = idx >> 21;
  size_t src = perm_src(np, tp, ep);
  qp[idx] = (_Float16)q[src];
  kp[idx] = (_Float16)k[src];
}

__global__ void pack_v_kernel(const float* __restrict__ v, _Float16* __restrict__ vpT) {
  int idx = blockIdx.x * 256 + threadIdx.x;         // [n'][e'][t'] flat (transposed)
  int tp = idx & (TT - 1);
  int ep = (idx >> 13) & 255;
  int np = idx >> 21;
  vpT[idx] = (_Float16)v[perm_src(np, tp, ep)];
}

// ---- flash attention over permuted tensors ------------------------------
__launch_bounds__(128)
__global__ void flash_kernel(const _Float16* __restrict__ qp, const _Float16* __restrict__ kp,
                             const _Float16* __restrict__ vpT, float* __restrict__ out) {
  // Double-buffered K / Vt tiles filled by GLOBAL_LOAD_ASYNC_TO_LDS (ASYNCcnt).
  __shared__ _Float16 sK[2][BK * SK_STR];      // 2 x 33792 B
  __shared__ _Float16 sVt[2][EE * SV_STR];     // 2 x 36864 B, [e][key]
  __shared__ _Float16 sP[4 * 16 * SV_STR];     // per-wave P scratch, 9216 B

  const int tid  = threadIdx.x;
  const int lane = tid & 31;
  const int w    = tid >> 5;
  const int hi16 = lane >> 4;    // 0: lanes 0-15, 1: lanes 16-31
  const int l15  = lane & 15;
  const int n    = blockIdx.y;                // permuted batch n'
  const int qbase = blockIdx.x * QROWS;

  // Q A-fragments for all 8 k-steps (kept in registers).
  // A k order used here (contiguous, same as B): lanes<16 hold K 0..15, lanes>=16 hold K 16..31.
  v16h qa[8];
  {
    const _Float16* qr = qp + (size_t)(n * TT + qbase + w * 16 + l15) * EE;
    #pragma unroll
    for (int s = 0; s < 8; ++s) {
      int base = s * 32 + hi16 * 16;
      qa[s] = cat16(*(const v8h*)(qr + base), *(const v8h*)(qr + base + 8));
    }
  }

  // Per-thread async-copy addressing: 16 x 16B chunks each for K and Vt.
  // K tile: 64 rows x 256 halves = 2048 chunks; Vt tile: 256 rows x 64 halves = 2048 chunks.
  const unsigned sKbase = (unsigned)(size_t)&sK[0][0];   // LDS byte address (low 32 bits of flat)
  const unsigned sVbase = (unsigned)(size_t)&sVt[0][0];
  unsigned kgB[16], vgB[16], klB[16], vlB[16];
  #pragma unroll
  for (int i = 0; i < 16; ++i) {
    int c = tid + 128 * i;                     // 0..2047
    int krow = c >> 5, kcol = (c & 31) * 8;
    int ve   = c >> 3, vcol = (c & 7) * 8;
    kgB[i] = (unsigned)(((unsigned)(n * TT + krow) * EE + kcol) * 2);
    vgB[i] = (unsigned)(((unsigned)(n * EE + ve) * TT + vcol) * 2);
    klB[i] = sKbase + (unsigned)((krow * SK_STR + kcol) * 2);
    vlB[i] = sVbase + (unsigned)((ve * SV_STR + vcol) * 2);
  }

  auto issue_tile = [&](int kv0, int buf) {
    const unsigned kvK = (unsigned)kv0 * (EE * 2);           // key row stride in K
    const unsigned kvV = (unsigned)kv0 * 2;                  // key col stride in Vt
    const unsigned kb  = buf ? (unsigned)(BK * SK_STR * 2) : 0u;
    const unsigned vb  = buf ? (unsigned)(EE * SV_STR * 2) : 0u;
    #pragma unroll
    for (int i = 0; i < 16; ++i) {
      asm volatile("global_load_async_to_lds_b128 %0, %1, %2"
                   :: "v"(klB[i] + kb), "v"(kgB[i] + kvK), "s"(kp) : "memory");
      asm volatile("global_load_async_to_lds_b128 %0, %1, %2"
                   :: "v"(vlB[i] + vb), "v"(vgB[i] + kvV), "s"(vpT) : "memory");
    }
  };

  v8f Oacc[16];
  #pragma unroll
  for (int j = 0; j < 16; ++j) Oacc[j] = (v8f){0.f,0.f,0.f,0.f,0.f,0.f,0.f,0.f};
  float mst[8], lst[8];
  #pragma unroll
  for (int vv = 0; vv < 8; ++vv) { mst[vv] = -__builtin_inff(); lst[vv] = 0.0f; }

  issue_tile(0, 0);                            // prologue: tile 0 -> buffer 0

  for (int it = 0; it < NIT; ++it) {
    asm volatile("s_wait_asynccnt 0x0" ::: "memory");   // my async stores to LDS done
    __syncthreads();                                    // everyone's tile ready
    issue_tile(((it + 1) * BK) & (TT - 1), (it + 1) & 1);  // overlap next tile with compute

    const _Float16* sKb = sK[it & 1];
    const _Float16* sVb = sVt[it & 1];

    // S = Q * K^T  (four 16-key column tiles)
    // B k order used here (contiguous, same as A): lane = column N; lanes<16 hold K 0..15, lanes>=16 hold K 16..31.
    v8f S[4];
    #pragma unroll
    for (int j = 0; j < 4; ++j) S[j] = (v8f){0.f,0.f,0.f,0.f,0.f,0.f,0.f,0.f};
    #pragma unroll
    for (int s = 0; s < 8; ++s) {
      int koff = s * 32 + hi16 * 16;
      #pragma unroll
      for (int j = 0; j < 4; ++j) {
        const _Float16* kr = sKb + (j * 16 + l15) * SK_STR + koff;
        v16h b = cat16(*(const v8h*)(kr), *(const v8h*)(kr + 8));
        S[j] = wmma_f16(qa[s], b, S[j]);
      }
    }

    // online softmax (C layout: row = slot + 8*hi16, col = l15; reduce over 16-lane groups)
    float alpha[8];
    _Float16* prw = sP + (w * 16) * SV_STR;
    #pragma unroll
    for (int vv = 0; vv < 8; ++vv) {
      float a0 = S[0][vv] * SCALE, a1 = S[1][vv] * SCALE;
      float a2 = S[2][vv] * SCALE, a3 = S[3][vv] * SCALE;
      float mx = fmaxf(fmaxf(a0, a1), fmaxf(a2, a3));
      mx = fmaxf(mx, __shfl_xor(mx, 1, 32));
      mx = fmaxf(mx, __shfl_xor(mx, 2, 32));
      mx = fmaxf(mx, __shfl_xor(mx, 4, 32));
      mx = fmaxf(mx, __shfl_xor(mx, 8, 32));
      float mnew = fmaxf(mst[vv], mx);
      float al = __expf(mst[vv] - mnew);
      float e0 = __expf(a0 - mnew);
      float e1 = __expf(a1 - mnew);
      float e2 = __expf(a2 - mnew);
      float e3 = __expf(a3 - mnew);
      float rs = (e0 + e1) + (e2 + e3);
      rs += __shfl_xor(rs, 1, 32);
      rs += __shfl_xor(rs, 2, 32);
      rs += __shfl_xor(rs, 4, 32);
      rs += __shfl_xor(rs, 8, 32);
      lst[vv] = lst[vv] * al + rs;
      mst[vv] = mnew;
      alpha[vv] = al;
      // P (f16) -> per-wave LDS scratch (re-layout C -> A fragment)
      int row = vv + hi16 * 8;
      prw[row * SV_STR +      l15] = (_Float16)e0;
      prw[row * SV_STR + 16 + l15] = (_Float16)e1;
      prw[row * SV_STR + 32 + l15] = (_Float16)e2;
      prw[row * SV_STR + 48 + l15] = (_Float16)e3;
    }
    v8f av;
    #pragma unroll
    for (int vv = 0; vv < 8; ++vv) av[vv] = alpha[vv];
    #pragma unroll
    for (int j = 0; j < 16; ++j) Oacc[j] *= av;

    // O += P * V   (two 32-deep k-steps; 16 output-column tiles)
    #pragma unroll
    for (int s2 = 0; s2 < 2; ++s2) {
      const _Float16* pr = sP + (w * 16 + l15) * SV_STR + s2 * 32 + hi16 * 16;
      v16h pa = cat16(*(const v8h*)(pr), *(const v8h*)(pr + 8));
      #pragma unroll
      for (int j = 0; j < 16; ++j) {
        const _Float16* vr = sVb + (j * 16 + l15) * SV_STR + s2 * 32 + hi16 * 16;
        v16h bv = cat16(*(const v8h*)(vr), *(const v8h*)(vr + 8));
        Oacc[j] = wmma_f16(pa, bv, Oacc[j]);
      }
    }
  }

  // epilogue: normalize + fused un-transpose scatter
  // a[n'][tau][ee] -> out[b>>3][(b&7)*1024 + ee*4 + n'*2 + (l2>>8)][l2&255], b=tau&15, l2=tau>>4
  #pragma unroll
  for (int vv = 0; vv < 8; ++vv) {
    float rl = 1.0f / lst[vv];
    int tau = qbase + w * 16 + vv + hi16 * 8;
    int b   = tau & 15;
    int l2  = tau >> 4;
    int npp = b >> 3;
    int tbase = (b & 7) * 1024 + n * 2 + (l2 >> 8);
    int epp = l2 & 255;
    size_t obase = (size_t)npp * ((size_t)TT * EE) + (size_t)epp;
    #pragma unroll
    for (int j = 0; j < 16; ++j) {
      int ee = j * 16 + l15;
      out[obase + (size_t)(tbase + ee * 4) * EE] = Oacc[j][vv] * rl;
    }
  }
}

extern "C" void kernel_launch(void* const* d_in, const int* in_sizes, int n_in,
                              void* d_out, int out_size, void* d_ws, size_t ws_size,
                              hipStream_t stream) {
  const float* q = (const float*)d_in[0];
  const float* k = (const float*)d_in[1];
  const float* v = (const float*)d_in[2];
  float* out = (float*)d_out;

  const size_t NE = (size_t)NB * TT * EE;            // 4,194,304 elements
  _Float16* qp  = (_Float16*)d_ws;                   // 8 MB
  _Float16* kp  = qp + NE;                           // 8 MB
  _Float16* vpT = kp + NE;                           // 8 MB (layout [n][e][t])

  const int threads = 256;
  const int blocksP = (int)(NE / threads);           // 16384
  pack_qk_kernel<<<blocksP, threads, 0, stream>>>(q, k, qp, kp);
  pack_v_kernel<<<blocksP, threads, 0, stream>>>(v, vpT);

  dim3 grid(TT / QROWS, NB);                         // (128, 2)
  flash_kernel<<<grid, 128, 0, stream>>>(qp, kp, vpT, out);
}
